// LSTM_10471130267946
// MI455X (gfx1250) — matrix operand order repaired and checked by reading, not compiled, not run on hardware
//
#include <hip/hip_runtime.h>
#include <stdint.h>

#define T_DIM 512
#define B_DIM 64
#define I_DIM 256
#define H_DIM 512
#define G_DIM 2048   // 4*H
#define LN_EPS 1e-5f

typedef __attribute__((ext_vector_type(16))) __bf16 v16bf;
typedef __attribute__((ext_vector_type(8)))  __bf16 v8bf;
typedef __attribute__((ext_vector_type(8)))  float  v8f;
typedef __attribute__((ext_vector_type(4)))  unsigned int v4u;
typedef __attribute__((ext_vector_type(8)))  int v8i_;
typedef __attribute__((ext_vector_type(4)))  int v4i_;

static __device__ __forceinline__ unsigned short f32_to_bf16_rne(float f) {
  union { float f; unsigned int u; } v; v.f = f;
  unsigned int u = v.u;
  return (unsigned short)((u + 0x7FFFu + ((u >> 16) & 1u)) >> 16);
}
static __device__ __forceinline__ float bf16_to_f32(unsigned short h) {
  union { unsigned int u; float f; } v; v.u = ((unsigned int)h) << 16;
  return v.f;
}
static __device__ __forceinline__ float sigmoidf_(float x) {
  return 1.0f / (1.0f + __expf(-x));
}

// ---------------------------------------------------------------------------
// Tensor Data Mover: 2-D tile load, global -> LDS.
// D# packing per CDNA5 ISA (groups 2/3 zero => <=2D).
//   group0: [1:0]=count=1 | lds_addr | global_addr[56:0] | type=2
//   group1: data_size | tensor_dim0/1 | tile_dim0/1 | tensor_dim0_stride
static __device__ __forceinline__ void tdm_load_2d(
    unsigned lds_addr, const void* gptr,
    unsigned tile_d0, unsigned tile_d1,
    unsigned tensor_d0, unsigned tensor_d1,
    unsigned stride_d0, unsigned data_size_log2 /*0:1B 1:2B 2:4B 3:8B*/) {
  unsigned long long ga = (unsigned long long)(uintptr_t)gptr;
  v4u g0;
  g0[0] = 1u;                                                   // count=1 (valid user D#)
  g0[1] = lds_addr;                                             // LDS byte address
  g0[2] = (unsigned)(ga & 0xFFFFFFFFu);                         // global_addr[31:0]
  g0[3] = (unsigned)((ga >> 32) & 0x01FFFFFFu) | (2u << 30);    // global_addr[56:32] | type=2
  v8i_ g1;
  g1[0] = (int)(data_size_log2 << 16);                          // wg_mask=0 | data_size
  g1[1] = (int)((tensor_d0 & 0xFFFFu) << 16);                   // tensor_dim0[15:0]
  g1[2] = (int)((tensor_d0 >> 16) & 0xFFFFu) |                  // tensor_dim0[31:16]
          (int)((tensor_d1 & 0xFFFFu) << 16);                   // tensor_dim1[15:0]
  g1[3] = (int)((tensor_d1 >> 16) & 0xFFFFu) |                  // tensor_dim1[31:16]
          (int)(tile_d0 << 16);                                 // tile_dim0
  g1[4] = (int)(tile_d1 & 0xFFFFu);                             // tile_dim1 (tile_dim2=0)
  g1[5] = (int)stride_d0;                                       // tensor_dim0_stride[31:0]
  g1[6] = 0;                                                    // stride hi / dim1_stride lo
  g1[7] = 0;
  v4i_ z4 = {0, 0, 0, 0};
#if defined(__clang_major__) && (__clang_major__ >= 23)
  v8i_ z8 = {0, 0, 0, 0, 0, 0, 0, 0};
  __builtin_amdgcn_tensor_load_to_lds(g0, g1, z4, z4, z8, 0);
#else
  __builtin_amdgcn_tensor_load_to_lds(g0, g1, z4, z4, 0);
#endif
}
static __device__ __forceinline__ void tdm_wait_all() {
  __builtin_amdgcn_s_wait_tensorcnt(0);
}
static __device__ __forceinline__ unsigned lds_offset_of(const void* p) {
  // LDS aperture decode truncates flat addr to addr[31:0] (ISA 10.2).
  return (unsigned)(uintptr_t)p;
}

// ---------------------------------------------------------------------------
// A fragment: 16x32 bf16 tile from row-major source (leading dim ld elements).
// Layout: lane L: M=L%16, K={kb..kb+15} contiguous (kb = 16*(lane>=16)).
static __device__ __forceinline__ v16bf load_a_frag(const __bf16* base, int ld, int lane) {
  const int m  = lane & 15;
  const int kb = (lane >> 4) << 4;  // 0 or 16
  const __bf16* p = base + m * ld;
  v8bf r0 = *(const v8bf*)(p + kb);
  v8bf r1 = *(const v8bf*)(p + 8 + kb);
  v16bf a;
#pragma unroll
  for (int e = 0; e < 8; ++e) { a[e] = r0[e]; a[8 + e] = r1[e]; }
  return a;
}

// B fragment: 32x16 (KxN), element (k,n) = W[n0+n][k0+k] with W row-major (ld elems).
// Lanes 0-15: N=lane, K=0..15 ; lanes 16-31: N=lane-16, K=16..31 (contiguous 32B run).
static __device__ __forceinline__ v16bf load_b_frag(const __bf16* W, int ld, int n0, int k0, int lane) {
  const int n  = lane & 15;
  const int ko = (lane >> 4) << 4;  // 0 or 16
  return *(const v16bf*)(W + (size_t)(n0 + n) * ld + k0 + ko);
}

static __device__ __forceinline__ v8f wmma_bf16(v16bf a, v16bf b, v8f c) {
  return __builtin_amdgcn_wmma_f32_16x16x32_bf16(false, a, false, b, (short)0, c, false, false);
}

// ---------------------------------------------------------------------------
__global__ void cvt_f32_to_bf16(const float* __restrict__ in,
                                unsigned short* __restrict__ out, int n) {
  int i = blockIdx.x * blockDim.x + threadIdx.x;
  if (i < n) out[i] = f32_to_bf16_rne(in[i]);
}

// ---------------------------------------------------------------------------
// Phase 1: gates_x = LN(xs @ Wxh^T + bxh) * gx + bx   ([32768,256]x[256,2048])
// One block = 16 rows, 256 threads (8 waves), wave covers 256 output cols.
// A tile staged to LDS by the Tensor Data Mover.
__global__ void gates_x_kernel(
    const unsigned short* __restrict__ xs_bf, const unsigned short* __restrict__ Wxh_bf,
    const float* __restrict__ bxh, const float* __restrict__ gx, const float* __restrict__ bx,
    float* __restrict__ gates_x)
{
  extern __shared__ char smem[];
  unsigned short* sA = (unsigned short*)smem;          // 16 x 256 bf16 (8 KB)
  float* sG = (float*)(smem + 16 * I_DIM * 2);         // 16 x 2048 f32 (128 KB)

  const int tid  = threadIdx.x;
  const int lane = tid & 31;
  const int wave = tid >> 5;                 // 0..7
  const size_t row0 = (size_t)blockIdx.x * 16;

  // TDM: 16x256 bf16 tile (contiguous rows of xs_bf) -> LDS
  if (wave == 0) {
    tdm_load_2d(lds_offset_of(sA), xs_bf + row0 * I_DIM,
                /*tile*/ I_DIM, 16,
                /*tensor*/ I_DIM, T_DIM * B_DIM,
                /*stride*/ I_DIM, /*2B*/ 1);
    tdm_wait_all();
  }
  __syncthreads();

  const __bf16* A  = (const __bf16*)sA;
  const __bf16* Wb = (const __bf16*)Wxh_bf;
  for (int half = 0; half < 2; ++half) {
    v8f acc[8];
#pragma unroll
    for (int nt = 0; nt < 8; ++nt)
#pragma unroll
      for (int e = 0; e < 8; ++e) acc[nt][e] = 0.0f;

    for (int kk = 0; kk < I_DIM / 32; ++kk) {
      v16bf a = load_a_frag(A + kk * 32, I_DIM, lane);
#pragma unroll
      for (int nt = 0; nt < 8; ++nt) {
        v16bf bb = load_b_frag(Wb, I_DIM, wave * 256 + half * 128 + nt * 16, kk * 32, lane);
        acc[nt] = wmma_bf16(a, bb, acc[nt]);
      }
    }
#pragma unroll
    for (int nt = 0; nt < 8; ++nt) {
      int col = wave * 256 + half * 128 + nt * 16 + (lane & 15);
      float bias = bxh[col];
#pragma unroll
      for (int e = 0; e < 8; ++e) {
        int r = e + ((lane >> 4) << 3);
        sG[r * G_DIM + col] = acc[nt][e] + bias;
      }
    }
  }
  __syncthreads();

  // LayerNorm over 2048 per row; 16 threads per row (half-wave butterflies).
  const int row = tid >> 4;
  const int t0  = tid & 15;
  float s = 0.0f, ss = 0.0f;
  for (int c2 = t0; c2 < G_DIM; c2 += 16) {
    float v = sG[row * G_DIM + c2];
    s += v; ss += v * v;
  }
#pragma unroll
  for (int off = 8; off >= 1; off >>= 1) {
    s  += __shfl_xor(s, off, 16);
    ss += __shfl_xor(ss, off, 16);
  }
  const float mu   = s * (1.0f / G_DIM);
  const float rstd = rsqrtf(ss * (1.0f / G_DIM) - mu * mu + LN_EPS);
  float* dst = gates_x + (row0 + row) * G_DIM;
  for (int c2 = t0; c2 < G_DIM; c2 += 16) {
    float v = sG[row * G_DIM + c2];
    dst[c2] = (v - mu) * rstd * gx[c2] + bx[c2];
  }
}

// ---------------------------------------------------------------------------
// Phase 2: sequential scan. 4 persistent blocks x 16 batch rows; 512 threads
// (16 waves); wave w == batch row w for all LN phases (pure wave32 shuffles).
// Per step, the TDM streams this step's 16x2048 gates_x tile into LDS while
// the waves run the recurrent WMMA GEMM.
__global__ void lstm_scan_kernel(
    const float* __restrict__ masks, const float* __restrict__ state,
    const unsigned short* __restrict__ Whh_bf, const float* __restrict__ bhh,
    const float* __restrict__ gh, const float* __restrict__ bh,
    const float* __restrict__ gc, const float* __restrict__ bc,
    const float* __restrict__ gates_x, float* __restrict__ out)
{
  extern __shared__ char smem[];
  unsigned short* sH = (unsigned short*)smem;                       // 16 x 512 bf16 ( 16 KB)
  float* sC  = (float*)(smem + 16 * H_DIM * 2);                     // 16 x 512 f32  ( 32 KB)
  float* sG  = (float*)(smem + 16 * H_DIM * 2 + 16 * H_DIM * 4);    // 16 x 2048 f32 (128 KB)
  float* sGX = (float*)(smem + 16 * H_DIM * 2 + 16 * H_DIM * 4
                             + 16 * G_DIM * 4);                     // 16 x 2048 f32 (128 KB)

  const int tid  = threadIdx.x;     // 0..511
  const int lane = tid & 31;
  const int wave = tid >> 5;        // 0..15
  const int row  = wave;            // batch row within group
  const int b    = blockIdx.x * 16 + row;

  // Warm L2 with Whh (streamed every step by all 4 blocks).
  {
    const char* wp = (const char*)Whh_bf;
    const size_t whh_bytes = (size_t)G_DIM * H_DIM * 2;
    for (size_t off = (size_t)tid * 64; off < whh_bytes; off += (size_t)512 * 64)
      __builtin_prefetch(wp + off, 0, 2);
  }

  // init h0 / c0 from state[B, 2H]
#pragma unroll
  for (int it = 0; it < 16; ++it) {
    int j = lane + (it << 5);
    sH[row * H_DIM + j] = f32_to_bf16_rne(state[(size_t)b * (2 * H_DIM) + j]);
    sC[row * H_DIM + j] = state[(size_t)b * (2 * H_DIM) + H_DIM + j];
  }

  const __bf16* Wb = (const __bf16*)Whh_bf;

  for (int t = 0; t < T_DIM; ++t) {
    // ---- A: issue TDM for this step's gates_x tile; apply episode-reset mask
    if (wave == 0) {
      tdm_load_2d(lds_offset_of(sGX),
                  gates_x + ((size_t)t * B_DIM + (size_t)blockIdx.x * 16) * G_DIM,
                  /*tile*/ G_DIM, 16,
                  /*tensor*/ G_DIM, T_DIM * B_DIM,
                  /*stride*/ G_DIM, /*4B*/ 2);
    }
    const float keep = 1.0f - masks[(size_t)t * B_DIM + b];
#pragma unroll
    for (int it = 0; it < 16; ++it) {
      int j = lane + (it << 5);
      sC[row * H_DIM + j] *= keep;
      sH[row * H_DIM + j] = f32_to_bf16_rne(bf16_to_f32(sH[row * H_DIM + j]) * keep);
    }
    __syncthreads();

    // ---- B: gates_h = h @ Whh^T + bhh  ([16,512]x[512,2048]); wave covers 128 cols
    {
      v8f acc[8];
#pragma unroll
      for (int nt = 0; nt < 8; ++nt)
#pragma unroll
        for (int e = 0; e < 8; ++e) acc[nt][e] = 0.0f;

      const __bf16* A = (const __bf16*)sH;
      for (int kk = 0; kk < H_DIM / 32; ++kk) {
        v16bf a = load_a_frag(A + kk * 32, H_DIM, lane);
#pragma unroll
        for (int nt = 0; nt < 8; ++nt) {
          v16bf bb = load_b_frag(Wb, H_DIM, wave * 128 + nt * 16, kk * 32, lane);
          acc[nt] = wmma_bf16(a, bb, acc[nt]);
        }
      }
#pragma unroll
      for (int nt = 0; nt < 8; ++nt) {
        int col = wave * 128 + nt * 16 + (lane & 15);
        float bias = bhh[col];
#pragma unroll
        for (int e = 0; e < 8; ++e) {
          int r = e + ((lane >> 4) << 3);
          sG[r * G_DIM + col] = acc[nt][e] + bias;
        }
      }
    }
    if (wave == 0) tdm_wait_all();   // TDM overlapped the GEMM; ensure tile landed
    __syncthreads();

    // ---- C: LN(gates_h) + gates_x(LDS), activations, c/h update (wave == row)
    {
      float s = 0.0f, ss = 0.0f;
      for (int c2 = lane; c2 < G_DIM; c2 += 32) {
        float v = sG[row * G_DIM + c2];
        s += v; ss += v * v;
      }
#pragma unroll
      for (int off = 16; off >= 1; off >>= 1) {
        s  += __shfl_xor(s, off, 32);
        ss += __shfl_xor(ss, off, 32);
      }
      const float mu   = s * (1.0f / G_DIM);
      const float rstd = rsqrtf(ss * (1.0f / G_DIM) - mu * mu + LN_EPS);

      const float* gxr = sGX + row * G_DIM;
      float cn[16], osg[16];
      float cs = 0.0f, css = 0.0f;
#pragma unroll
      for (int it = 0; it < 16; ++it) {
        int j = lane + (it << 5);
        float fg = (sG[row*G_DIM + j           ] - mu) * rstd * gh[j           ] + bh[j           ] + gxr[j           ];
        float ig = (sG[row*G_DIM + H_DIM + j   ] - mu) * rstd * gh[H_DIM + j   ] + bh[H_DIM + j   ] + gxr[H_DIM + j   ];
        float cg = (sG[row*G_DIM + 2*H_DIM + j ] - mu) * rstd * gh[2*H_DIM + j ] + bh[2*H_DIM + j ] + gxr[2*H_DIM + j ];
        float og = (sG[row*G_DIM + 3*H_DIM + j ] - mu) * rstd * gh[3*H_DIM + j ] + bh[3*H_DIM + j ] + gxr[3*H_DIM + j ];
        float c_new = sigmoidf_(fg) * sC[row * H_DIM + j] + sigmoidf_(ig) * tanhf(cg);
        sC[row * H_DIM + j] = c_new;
        cn[it]  = c_new;
        osg[it] = sigmoidf_(og);
        cs += c_new; css += c_new * c_new;
      }
#pragma unroll
      for (int off = 16; off >= 1; off >>= 1) {
        cs  += __shfl_xor(cs, off, 32);
        css += __shfl_xor(css, off, 32);
      }
      const float muc   = cs * (1.0f / H_DIM);
      const float rstdc = rsqrtf(css * (1.0f / H_DIM) - muc * muc + LN_EPS);

      float* hs = out + ((size_t)t * B_DIM + b) * H_DIM;
#pragma unroll
      for (int it = 0; it < 16; ++it) {
        int j = lane + (it << 5);
        float lnc = (cn[it] - muc) * rstdc * gc[j] + bc[j];
        float h = osg[it] * tanhf(lnc);
        hs[j] = h;
        sH[row * H_DIM + j] = f32_to_bf16_rne(h);
        if (t == T_DIM - 1) {
          float* hc = out + (size_t)T_DIM * B_DIM * H_DIM + (size_t)b * (2 * H_DIM);
          hc[j]          = h;
          hc[H_DIM + j]  = cn[it];
        }
      }
    }
    __syncthreads();
  }
}

// ---------------------------------------------------------------------------
extern "C" void kernel_launch(void* const* d_in, const int* in_sizes, int n_in,
                              void* d_out, int out_size, void* d_ws, size_t ws_size,
                              hipStream_t stream) {
  (void)in_sizes; (void)n_in; (void)out_size; (void)ws_size;
  const float* xs    = (const float*)d_in[0];
  const float* masks = (const float*)d_in[1];
  const float* state = (const float*)d_in[2];
  const float* Wxh   = (const float*)d_in[3];
  const float* bxh   = (const float*)d_in[4];
  const float* Whh   = (const float*)d_in[5];
  const float* bhh   = (const float*)d_in[6];
  const float* gx    = (const float*)d_in[7];
  const float* bx    = (const float*)d_in[8];
  const float* gh    = (const float*)d_in[9];
  const float* bh    = (const float*)d_in[10];
  const float* gc    = (const float*)d_in[11];
  const float* bc    = (const float*)d_in[12];
  float* out = (float*)d_out;

  uint8_t* ws = (uint8_t*)d_ws;
  const size_t xs_elems  = (size_t)T_DIM * B_DIM * I_DIM;   // 8,388,608
  const size_t wxh_elems = (size_t)G_DIM * I_DIM;           // 524,288
  const size_t whh_elems = (size_t)G_DIM * H_DIM;           // 1,048,576
  unsigned short* xs_bf  = (unsigned short*)ws;
  unsigned short* Wxh_bf = (unsigned short*)(ws + 2 * xs_elems);
  unsigned short* Whh_bf = (unsigned short*)(ws + 2 * (xs_elems + wxh_elems));
  float* gates_x = (float*)(ws + 2 * (xs_elems + wxh_elems + whh_elems));

  cvt_f32_to_bf16<<<(int)((xs_elems  + 255) / 256), 256, 0, stream>>>(xs,  xs_bf,  (int)xs_elems);
  cvt_f32_to_bf16<<<(int)((wxh_elems + 255) / 256), 256, 0, stream>>>(Wxh, Wxh_bf, (int)wxh_elems);
  cvt_f32_to_bf16<<<(int)((whh_elems + 255) / 256), 256, 0, stream>>>(Whh, Whh_bf, (int)whh_elems);

  const size_t lds1 = (size_t)16 * I_DIM * 2 + (size_t)16 * G_DIM * 4;  // 136 KB
  gates_x_kernel<<<(T_DIM * B_DIM) / 16, 256, lds1, stream>>>(xs_bf, Wxh_bf, bxh, gx, bx, gates_x);

  const size_t lds2 = (size_t)16 * H_DIM * 2 + (size_t)16 * H_DIM * 4 +
                      (size_t)2 * 16 * G_DIM * 4;                       // 304 KB
  lstm_scan_kernel<<<B_DIM / 16, 512, lds2, stream>>>(masks, state, Whh_bf, bhh,
                                                      gh, bh, gc, bc, gates_x, out);
}
